// GAT_mlp_61976378081536
// MI455X (gfx1250) — matrix operand order repaired and checked by reading, not compiled, not run on hardware
//
#include <hip/hip_runtime.h>
#include <hip/hip_bf16.h>

// ---------------------------------------------------------------------------
// Types for CDNA5 WMMA (gfx1250, wave32)
// ---------------------------------------------------------------------------
typedef __attribute__((ext_vector_type(16))) __bf16 v16bf;
typedef __attribute__((ext_vector_type(8)))  float  v8f;

union Frag16 { v16bf v; unsigned u[8]; uint4 q[2]; };

__device__ __forceinline__ unsigned short f2bf(float f) {
  unsigned a = __float_as_uint(f);
  a = (a + 0x7FFFu + ((a >> 16) & 1u)) >> 16;   // RNE fp32 -> bf16
  return (unsigned short)a;
}

// ---------------------------------------------------------------------------
// One-time converts (tiny): W[K,M] fp32 -> Wt[M,K] bf16 ; and row-major fp32->bf16
// ---------------------------------------------------------------------------
__global__ void w_to_bf16t_kernel(const float* __restrict__ W, unsigned short* __restrict__ Wt,
                                  int K, int M) {
  int i = blockIdx.x * blockDim.x + threadIdx.x;
  if (i >= K * M) return;
  const int k = i / M, m = i % M;
  Wt[(size_t)m * K + k] = f2bf(W[i]);
}

__global__ void to_bf16_kernel(const float* __restrict__ s, unsigned short* __restrict__ d,
                               long long n) {
  long long i = (long long)blockIdx.x * blockDim.x + threadIdx.x;
  if (i < n) d[i] = f2bf(s[i]);
}

// ---------------------------------------------------------------------------
// WMMA GEMM: C[N,M] = epilogue(A[N,K]bf16 @ Wt[M,K]bf16^T)
//   EPI=0: none   EPI=1: +bias   EPI=2: +bias, BN(eval), ReLU
// Wave: 16 rows x 64 cols (4 tiles); K fully unrolled (KSTEPS x 32).
// Fragments loaded as b128 per ISA 7.12.2 layouts. Optional bf16 copy of C.
// ---------------------------------------------------------------------------
template <int EPI, int KSTEPS>
__global__ __launch_bounds__(256) void wmma_gemm_kernel(
    const unsigned short* __restrict__ A,   // [N,K] bf16 row-major
    const unsigned short* __restrict__ Wt,  // [M,K] bf16 (W transposed)
    const float* __restrict__ bias,
    const float* __restrict__ bng, const float* __restrict__ bnb,
    const float* __restrict__ bnm, const float* __restrict__ bnv,
    float* __restrict__ C, unsigned short* __restrict__ Cbf,
    int N, int M) {
  constexpr int K = KSTEPS * 32;
  const int lane = threadIdx.x & 31;
  const int wid  = threadIdx.x >> 5;
  const int half = lane >> 4;
  const int l16  = lane & 15;

  const int rowBase = (blockIdx.x * 8 + wid) * 16;
  if (rowBase >= N) return;                 // wave-uniform (EXEC stays full)
  const int colBase = blockIdx.y * 64;

  v8f acc[4];
#pragma unroll
  for (int t = 0; t < 4; ++t)
#pragma unroll
    for (int j = 0; j < 8; ++j) acc[t][j] = 0.0f;

  int ar = rowBase + l16; if (ar > N - 1) ar = N - 1;
  const unsigned short* Arow = A + (size_t)ar * K;

  const unsigned short* Brow[4];
#pragma unroll
  for (int t = 0; t < 4; ++t) {
    int c = colBase + t * 16 + l16; if (c > M - 1) c = M - 1;
    Brow[t] = Wt + (size_t)c * K;
  }

#pragma unroll
  for (int ks = 0; ks < KSTEPS; ++ks) {
    const int kb = ks * 32;
    // A fragment 16x32: lane needs k = kb + 8*half + [0..7] and kb+16+8*half+[0..7]
    Frag16 a;
    a.q[0] = *(const uint4*)(Arow + kb + half * 8);
    a.q[1] = *(const uint4*)(Arow + kb + 16 + half * 8);
#pragma unroll
    for (int t = 0; t < 4; ++t) {
      // B fragment 32x16: lane needs k = kb + 8*half + [0..7] and kb+16+8*half+[0..7] of its column
      Frag16 b;
      b.q[0] = *(const uint4*)(Brow[t] + kb + half * 8);
      b.q[1] = *(const uint4*)(Brow[t] + kb + 16 + half * 8);
      acc[t] = __builtin_amdgcn_wmma_f32_16x16x32_bf16(
          false, a.v, false, b.v, (short)0, acc[t], false, false);
    }
  }

  // Epilogue + store. C layout: VGPR j -> row = rowBase + j + 8*half, col = lane&15.
#pragma unroll
  for (int t = 0; t < 4; ++t) {
    const int c = colBase + t * 16 + l16;
    if (c >= M) continue;
    float add = 0.0f, scale = 1.0f, shift = 0.0f;
    if (EPI >= 1) add = bias[c];
    if (EPI == 2) {
      const float g = bng[c], be = bnb[c], mu = bnm[c], va = bnv[c];
      scale = g * rsqrtf(va + 1e-5f);
      shift = be - mu * scale;
    }
#pragma unroll
    for (int j = 0; j < 8; ++j) {
      const int r = rowBase + j + (half << 3);
      if (r < N) {
        float v = acc[t][j];
        if (EPI >= 1) v += add;
        if (EPI == 2) { v = v * scale + shift; v = fmaxf(v, 0.0f); }
        C[(size_t)r * M + c] = v;
        if (Cbf) Cbf[(size_t)r * M + c] = f2bf(v);
      }
    }
  }
}

// ---------------------------------------------------------------------------
__global__ void fill_kernel(float* __restrict__ p, float v, long long n) {
  long long i = (long long)blockIdx.x * blockDim.x + threadIdx.x;
  if (i < n) p[i] = v;
}

// el[n,h] = sum_d z[n,h*64+d]*al[h*64+d]; er likewise. Thread per (node,head).
__global__ void el_er_kernel(const float* __restrict__ z,
                             const float* __restrict__ al,
                             const float* __restrict__ ar,
                             float* __restrict__ el, float* __restrict__ er, int N) {
  int t = blockIdx.x * blockDim.x + threadIdx.x;
  if (t >= N * 4) return;
  const int n = t >> 2, h = t & 3;
  const float4* zr  = (const float4*)(z  + (size_t)n * 256 + h * 64);
  const float4* alr = (const float4*)(al + h * 64);
  const float4* arr = (const float4*)(ar + h * 64);
  float sl = 0.0f, sr = 0.0f;
#pragma unroll 4
  for (int d = 0; d < 16; ++d) {
    const float4 zv = zr[d], a4 = alr[d], r4 = arr[d];
    sl += zv.x * a4.x + zv.y * a4.y + zv.z * a4.z + zv.w * a4.w;
    sr += zv.x * r4.x + zv.y * r4.y + zv.z * r4.z + zv.w * r4.w;
  }
  el[t] = sl; er[t] = sr;
}

// Atomic float max via int/uint atomics (handles negatives; init to -inf)
__device__ __forceinline__ void atomicMaxFloat(float* addr, float v) {
  if (v >= 0.0f) atomicMax((int*)addr, __float_as_int(v));
  else           atomicMin((unsigned int*)addr, __float_as_uint(v));
}

__device__ __forceinline__ float leaky02(float v) { return v > 0.0f ? v : 0.2f * v; }

__global__ void edge_max_kernel(const int* __restrict__ src, const int* __restrict__ dst,
                                const float* __restrict__ el, const float* __restrict__ er,
                                float* __restrict__ emax, int E) {
  int t = blockIdx.x * blockDim.x + threadIdx.x;
  if (t >= E * 4) return;
  const int e = t >> 2, h = t & 3;
  const int s = src[e], d = dst[e];
  const float v = leaky02(el[s * 4 + h] + er[d * 4 + h]);
  atomicMaxFloat(&emax[d * 4 + h], v);
}

__global__ void edge_exp_kernel(const int* __restrict__ src, const int* __restrict__ dst,
                                const float* __restrict__ el, const float* __restrict__ er,
                                const float* __restrict__ emax,
                                float* __restrict__ ew, float* __restrict__ denom, int E) {
  int t = blockIdx.x * blockDim.x + threadIdx.x;
  if (t >= E * 4) return;
  const int e = t >> 2, h = t & 3;
  const int s = src[e], d = dst[e];
  const float v = leaky02(el[s * 4 + h] + er[d * 4 + h]);
  const float w = __expf(v - emax[d * 4 + h]);
  ew[t] = w;
  atomicAdd(&denom[d * 4 + h], w);
}

// agg[dst,c] += (ew/denom[dst]) * z[src,c]. Block = 64 threads, 1 edge, float4 loads.
__global__ void edge_scatter_kernel(const int* __restrict__ src, const int* __restrict__ dst,
                                    const float* __restrict__ z, const float* __restrict__ ew,
                                    const float* __restrict__ denom, float* __restrict__ agg,
                                    int E) {
  const int e = blockIdx.x;
  if (e >= E) return;
  const int c4 = threadIdx.x;         // 0..63 -> cols [c4*4, c4*4+4)
  const int h  = c4 >> 4;
  const int s = src[e], d = dst[e];
  const float alpha = ew[e * 4 + h] / denom[d * 4 + h];
  const float4 zv = ((const float4*)(z + (size_t)s * 256))[c4];
  float* ap = agg + (size_t)d * 256 + c4 * 4;
  atomicAdd(ap + 0, alpha * zv.x);
  atomicAdd(ap + 1, alpha * zv.y);
  atomicAdd(ap + 2, alpha * zv.z);
  atomicAdd(ap + 3, alpha * zv.w);
}

// out = elu(agg + conv_b[col] + skip); writes fp32 and bf16 copies
__global__ void combine_elu_kernel(const float* __restrict__ agg, const float* __restrict__ cb,
                                   const float* __restrict__ skip, float* __restrict__ out,
                                   unsigned short* __restrict__ out_bf, long long total) {
  long long i = (long long)blockIdx.x * blockDim.x + threadIdx.x;
  if (i >= total) return;
  const int col = (int)(i & 255);
  const float v = agg[i] + cb[col] + skip[i];
  const float r = v > 0.0f ? v : (__expf(v) - 1.0f);
  out[i] = r;
  out_bf[i] = f2bf(r);
}

// log_softmax over M<=64 classes, one wave32 per row (each lane covers 2 cols)
__global__ void log_softmax_kernel(const float* __restrict__ logits, float* __restrict__ out,
                                   int N, int M) {
  const int row  = blockIdx.x * 8 + (threadIdx.x >> 5);
  const int lane = threadIdx.x & 31;
  if (row >= N) return;
  const float* r = logits + (size_t)row * M;
  const float v0 = (lane      < M) ? r[lane]      : -__builtin_inff();
  const float v1 = (lane + 32 < M) ? r[lane + 32] : -__builtin_inff();
  float m = fmaxf(v0, v1);
#pragma unroll
  for (int o = 16; o > 0; o >>= 1) m = fmaxf(m, __shfl_xor(m, o, 32));
  float s = (lane      < M ? __expf(v0 - m) : 0.0f) +
            (lane + 32 < M ? __expf(v1 - m) : 0.0f);
#pragma unroll
  for (int o = 16; o > 0; o >>= 1) s += __shfl_xor(s, o, 32);
  const float ls = __logf(s) + m;
  if (lane      < M) out[(size_t)row * M + lane]      = v0 - ls;
  if (lane + 32 < M) out[(size_t)row * M + lane + 32] = v1 - ls;
}

// ---------------------------------------------------------------------------
// Host-side helpers
// ---------------------------------------------------------------------------
static void launch_gemm(int epi, const unsigned short* A, const unsigned short* Wt,
                        const float* bias, const float* g, const float* b,
                        const float* mu, const float* va,
                        float* C, unsigned short* Cbf, int N, int K, int M, hipStream_t s) {
  const int rowTiles = (N + 15) / 16;
  dim3 grid((rowTiles + 7) / 8, (M + 63) / 64);
  dim3 blk(256);
  if (K == 128) {
    if (epi == 0)      wmma_gemm_kernel<0, 4><<<grid, blk, 0, s>>>(A, Wt, bias, g, b, mu, va, C, Cbf, N, M);
    else if (epi == 1) wmma_gemm_kernel<1, 4><<<grid, blk, 0, s>>>(A, Wt, bias, g, b, mu, va, C, Cbf, N, M);
    else               wmma_gemm_kernel<2, 4><<<grid, blk, 0, s>>>(A, Wt, bias, g, b, mu, va, C, Cbf, N, M);
  } else { // K == 256
    if (epi == 0)      wmma_gemm_kernel<0, 8><<<grid, blk, 0, s>>>(A, Wt, bias, g, b, mu, va, C, Cbf, N, M);
    else if (epi == 1) wmma_gemm_kernel<1, 8><<<grid, blk, 0, s>>>(A, Wt, bias, g, b, mu, va, C, Cbf, N, M);
    else               wmma_gemm_kernel<2, 8><<<grid, blk, 0, s>>>(A, Wt, bias, g, b, mu, va, C, Cbf, N, M);
  }
}

static void run_gat_edges(const int* src, const int* dst, const float* z,
                          const float* al, const float* ar,
                          float* el, float* er, float* emax, float* denom, float* ew,
                          float* agg, int N, int E, hipStream_t s) {
  const long long nh = (long long)N * 256;
  fill_kernel<<<(unsigned)((nh + 255) / 256), 256, 0, s>>>(agg, 0.0f, nh);
  fill_kernel<<<(unsigned)((N * 4 + 255) / 256), 256, 0, s>>>(emax, -__builtin_huge_valf(), (long long)N * 4);
  fill_kernel<<<(unsigned)((N * 4 + 255) / 256), 256, 0, s>>>(denom, 0.0f, (long long)N * 4);
  el_er_kernel<<<(N * 4 + 255) / 256, 256, 0, s>>>(z, al, ar, el, er, N);
  edge_max_kernel<<<(E * 4 + 255) / 256, 256, 0, s>>>(src, dst, el, er, emax, E);
  edge_exp_kernel<<<(E * 4 + 255) / 256, 256, 0, s>>>(src, dst, el, er, emax, ew, denom, E);
  edge_scatter_kernel<<<E, 64, 0, s>>>(src, dst, z, ew, denom, agg, E);
}

extern "C" void kernel_launch(void* const* d_in, const int* in_sizes, int n_in,
                              void* d_out, int out_size, void* d_ws, size_t ws_size,
                              hipStream_t stream) {
  const float* x        = (const float*)d_in[0];
  const float* conv0_W  = (const float*)d_in[1];
  const float* conv0_al = (const float*)d_in[2];
  const float* conv0_ar = (const float*)d_in[3];
  const float* conv0_b  = (const float*)d_in[4];
  const float* skip0_W  = (const float*)d_in[5];
  const float* skip0_b  = (const float*)d_in[6];
  const float* conv1_W  = (const float*)d_in[7];
  const float* conv1_al = (const float*)d_in[8];
  const float* conv1_ar = (const float*)d_in[9];
  const float* conv1_b  = (const float*)d_in[10];
  const float* skip1_W  = (const float*)d_in[11];
  const float* skip1_b  = (const float*)d_in[12];
  const float* mlp_W1   = (const float*)d_in[13];
  const float* mlp_b1   = (const float*)d_in[14];
  const float* bn_g     = (const float*)d_in[15];
  const float* bn_b     = (const float*)d_in[16];
  const float* bn_m     = (const float*)d_in[17];
  const float* bn_v     = (const float*)d_in[18];
  const float* mlp_W2   = (const float*)d_in[19];
  const float* mlp_b2   = (const float*)d_in[20];
  const int*   esrc     = (const int*)d_in[21];
  const int*   edst     = (const int*)d_in[22];

  const int N = in_sizes[0] / 128;   // 50000
  const int E = in_sizes[21];        // 800000
  const int C = in_sizes[20];        // 47 classes

  // ---- workspace layout ----
  float* ws = (float*)d_ws;
  const size_t nh = (size_t)N * 256;
  float* Bz    = ws;                 // z of current layer (fp32)
  float* Bs    = Bz + nh;            // skip / logits reuse (fp32)
  float* Bagg  = Bs + nh;            // edge aggregation (fp32)
  float* Bh    = Bagg + nh;          // layer output h (fp32)
  float* el    = Bh + nh;
  float* er    = el + (size_t)N * 4;
  float* emax  = er + (size_t)N * 4;
  float* denom = emax + (size_t)N * 4;
  float* ew    = denom + (size_t)N * 4;
  float* bf_area = ew + (size_t)E * 4;
  // bf16 buffers carved as ushort
  unsigned short* xbf = (unsigned short*)bf_area;            // N*128
  unsigned short* hbf = xbf + (size_t)N * 128;               // N*256
  unsigned short* mbf = hbf + (size_t)N * 256;               // N*256
  unsigned short* wt0 = mbf + (size_t)N * 256;               // conv0_Wt 256x128
  unsigned short* wts0 = wt0 + 256 * 128;                    // skip0_Wt 256x128
  unsigned short* wt1 = wts0 + 256 * 128;                    // conv1_Wt 256x256
  unsigned short* wts1 = wt1 + 256 * 256;                    // skip1_Wt 256x256
  unsigned short* wtm1 = wts1 + 256 * 256;                   // mlp_W1t 256x256
  unsigned short* wtm2 = wtm1 + 256 * 256;                   // mlp_W2t 47x256
  const long long tot = (long long)nh;

  // ---- one-time converts (tiny) ----
  w_to_bf16t_kernel<<<(128 * 256 + 255) / 256, 256, 0, stream>>>(conv0_W, wt0, 128, 256);
  w_to_bf16t_kernel<<<(128 * 256 + 255) / 256, 256, 0, stream>>>(skip0_W, wts0, 128, 256);
  w_to_bf16t_kernel<<<(256 * 256 + 255) / 256, 256, 0, stream>>>(conv1_W, wt1, 256, 256);
  w_to_bf16t_kernel<<<(256 * 256 + 255) / 256, 256, 0, stream>>>(skip1_W, wts1, 256, 256);
  w_to_bf16t_kernel<<<(256 * 256 + 255) / 256, 256, 0, stream>>>(mlp_W1, wtm1, 256, 256);
  w_to_bf16t_kernel<<<(256 * C + 255) / 256, 256, 0, stream>>>(mlp_W2, wtm2, 256, C);
  to_bf16_kernel<<<(unsigned)(((long long)N * 128 + 255) / 256), 256, 0, stream>>>(x, xbf, (long long)N * 128);

  // ---- Layer 0 (K=128) ----
  launch_gemm(0, xbf, wt0, nullptr, nullptr, nullptr, nullptr, nullptr, Bz, nullptr, N, 128, 256, stream);
  launch_gemm(1, xbf, wts0, skip0_b, nullptr, nullptr, nullptr, nullptr, Bs, nullptr, N, 128, 256, stream);
  run_gat_edges(esrc, edst, Bz, conv0_al, conv0_ar, el, er, emax, denom, ew, Bagg, N, E, stream);
  combine_elu_kernel<<<(unsigned)((tot + 255) / 256), 256, 0, stream>>>(Bagg, conv0_b, Bs, Bh, hbf, tot);

  // ---- Layer 1 (K=256) ----
  launch_gemm(0, hbf, wt1, nullptr, nullptr, nullptr, nullptr, nullptr, Bz, nullptr, N, 256, 256, stream);
  launch_gemm(1, hbf, wts1, skip1_b, nullptr, nullptr, nullptr, nullptr, Bs, nullptr, N, 256, 256, stream);
  run_gat_edges(esrc, edst, Bz, conv1_al, conv1_ar, el, er, emax, denom, ew, Bagg, N, E, stream);
  combine_elu_kernel<<<(unsigned)((tot + 255) / 256), 256, 0, stream>>>(Bagg, conv1_b, Bs, Bh, hbf, tot);

  // ---- MLP head ----
  launch_gemm(2, hbf, wtm1, mlp_b1, bn_g, bn_b, bn_m, bn_v, Bz, mbf, N, 256, 256, stream); // BN+ReLU fused, bf16 copy
  launch_gemm(1, mbf, wtm2, mlp_b2, nullptr, nullptr, nullptr, nullptr, Bs, nullptr, N, 256, C, stream);
  log_softmax_kernel<<<(N + 7) / 8, 256, 0, stream>>>(Bs, (float*)d_out, N, C);
}
